// Bi_Encoder_42528766165268
// MI455X (gfx1250) — compile-verified, matrix operand order checked by reading, not run on hardware
//
#include <hip/hip_runtime.h>
#include <hip/hip_bf16.h>
#include <math.h>

typedef __bf16 bf16;
typedef __attribute__((ext_vector_type(8)))  bf16  v8bf;
typedef __attribute__((ext_vector_type(16))) bf16  v16bf;
typedef __attribute__((ext_vector_type(8)))  float v8f;

#define B_  64
#define S_  512
#define E_  512
#define H_  1024
#define H2_ 512

// ---------------------------------------------------------------------------
// Fragment loader for v_wmma_f32_16x16x32_bf16 (wave32).
// 16-bit A-matrix layout (ISA 7.12.2): lanes 0-15 hold row M=lane,
// K = {0..7} packed in VGPR0-3 and {16..23} in VGPR4-7; lanes 16-31 hold the
// same row with K = {8..15} and {24..31}.  With row-major (ld=K) data this is
// two contiguous 16-byte loads per lane (verified in round-1 asm as paired
// global_load_b128 at +0/+32).  B-frags use the same loader on N x K
// (pre-transposed) weights.
// ---------------------------------------------------------------------------
__device__ __forceinline__ v16bf load_frag_k32(const bf16* __restrict__ base,
                                               int row_base, int ld, int k0) {
  const int lane = threadIdx.x & 31;
  const bf16* p = base + (size_t)(row_base + (lane & 15)) * ld + k0 + ((lane >> 4) << 3);
  v8bf lo = *(const v8bf*)(p);        // K = k0 +  0..7   (or 8..15 for hi lanes)
  v8bf hi = *(const v8bf*)(p + 16);   // K = k0 + 16..23  (or 24..31)
  v16bf f;
#pragma unroll
  for (int i = 0; i < 8; ++i) { f[i] = lo[i]; f[i + 8] = hi[i]; }
  return f;
}

// Same fragment gather but from an LDS-resident tile (emits ds_load_b128).
__device__ __forceinline__ v16bf load_frag_lds(const bf16* base,
                                               int row_base, int ld, int k0) {
  const int lane = threadIdx.x & 31;
  const bf16* p = base + (size_t)(row_base + (lane & 15)) * ld + k0 + ((lane >> 4) << 3);
  v8bf lo = *(const v8bf*)(p);
  v8bf hi = *(const v8bf*)(p + 16);
  v16bf f;
#pragma unroll
  for (int i = 0; i < 8; ++i) { f[i] = lo[i]; f[i + 8] = hi[i]; }
  return f;
}

// ---------------------------------------------------------------------------
// Grid-wide sense-reversing barrier (L2 atomics + s_sleep spin).
// cnt/gen live in d_ws and are zeroed by init_state_kernel every launch.
// ---------------------------------------------------------------------------
__device__ __forceinline__ void grid_barrier(unsigned* cnt, unsigned* gen_p,
                                             unsigned nblocks) {
  __threadfence();
  __syncthreads();
  if (threadIdx.x == 0) {
    volatile unsigned* gen = (volatile unsigned*)gen_p;
    unsigned g = *gen;
    if (atomicAdd(cnt, 1u) == nblocks - 1u) {
      *cnt = 0u;
      __threadfence();
      *gen = g + 1u;
    } else {
      while (*gen == g) { __builtin_amdgcn_s_sleep(2); }
    }
    __threadfence();
  }
  __syncthreads();
}

// ---------------------------------------------------------------------------
// Small utility kernels
// ---------------------------------------------------------------------------
__global__ void cast_f32_bf16_kernel(const float* __restrict__ in,
                                     bf16* __restrict__ out, size_t n) {
  size_t stride = (size_t)gridDim.x * blockDim.x;
  for (size_t i = (size_t)blockIdx.x * blockDim.x + threadIdx.x; i < n; i += stride)
    out[i] = (bf16)in[i];
}

// W is (K x N) row-major fp32; Wt is (N x K) row-major bf16.
__global__ void transpose_cast_kernel(const float* __restrict__ W,
                                      bf16* __restrict__ Wt, int K, int N) {
  size_t total = (size_t)K * N;
  size_t stride = (size_t)gridDim.x * blockDim.x;
  for (size_t i = (size_t)blockIdx.x * blockDim.x + threadIdx.x; i < total; i += stride) {
    size_t n = i / (size_t)K;
    size_t k = i - n * (size_t)K;
    Wt[i] = (bf16)W[k * (size_t)N + n];
  }
}

__global__ void init_state_kernel(float* h_f, float* h_b,
                                  bf16* hbf_f, bf16* hbf_b, unsigned* bar) {
  int stride = gridDim.x * blockDim.x;
  for (int j = blockIdx.x * blockDim.x + threadIdx.x; j < B_ * H_; j += stride) {
    h_f[j] = 0.f;  h_b[j] = 0.f;
    hbf_f[j] = (bf16)0.f;  hbf_b[j] = (bf16)0.f;
  }
  if (blockIdx.x == 0 && threadIdx.x == 0) { bar[0] = 0u; bar[1] = 0u; }
}

// ---------------------------------------------------------------------------
// Generic bf16 WMMA GEMM:  C = act(A[MxK] @ Bt[NxK]^T + bias)
// MODE 0: store bf16 row-major (M x N)          (input-MLP hidden layer)
// MODE 1: store fp32 to xh[(s*B + b)*H + n]     (row m = b*S + s transpose)
// Block = 256 threads = 8 waves; block tile 64(M) x 128(N); each wave does a
// 16x64 strip: one A-frag reused across 4 WMMAs per 32-wide K chunk.
// ---------------------------------------------------------------------------
template<int MODE, bool RELU>
__global__ void __launch_bounds__(256)
gemm_bf16_kernel(const bf16* __restrict__ A, const bf16* __restrict__ Bt,
                 const float* __restrict__ bias, void* __restrict__ out,
                 int M, int N, int K) {
  const int wave = threadIdx.x >> 5;
  const int lane = threadIdx.x & 31;
  const int m_tile = blockIdx.y * 64 + (wave & 3) * 16;
  const int n_base = blockIdx.x * 128 + (wave >> 2) * 64;

  v8f acc[4] = {};
  for (int k0 = 0; k0 < K; k0 += 32) {
    v16bf a = load_frag_k32(A, m_tile, K, k0);
    if (k0 + 32 < K)  // warm next A chunk into cache: global_prefetch_b8
      __builtin_prefetch(A + (size_t)(m_tile + (lane & 15)) * K + k0 + 32, 0, 1);
#pragma unroll
    for (int j = 0; j < 4; ++j) {
      v16bf w = load_frag_k32(Bt, n_base + j * 16, K, k0);
      acc[j] = __builtin_amdgcn_wmma_f32_16x16x32_bf16(
          false, a, false, w, (short)0, acc[j], false, false);
    }
  }

  const int n_local = lane & 15;
  const int m_off = (lane >> 4) << 3;  // lanes 16-31 hold rows M = 8..15 of tile
#pragma unroll
  for (int j = 0; j < 4; ++j) {
    const int n = n_base + j * 16 + n_local;
    const float bv = bias[n];
#pragma unroll
    for (int r = 0; r < 8; ++r) {
      const int m = m_tile + m_off + r;
      float v = acc[j][r] + bv;
      if (RELU) v = v > 0.f ? v : 0.f;
      if (MODE == 0) {
        ((bf16*)out)[(size_t)m * N + n] = (bf16)v;
      } else {
        const int bb = m >> 9;        // m / S
        const int ss = m & (S_ - 1);  // m % S
        ((float*)out)[((size_t)ss * B_ + bb) * H_ + n] = v;
      }
    }
  }
}

// ---------------------------------------------------------------------------
// Persistent bidirectional scan.  64 blocks x 256 threads; even blocks = fwd,
// odd = bwd (32 blocks = 256 waves per direction).  Per step:
//   layer1: mid(64x512)  = relu(hbf @ Whh1 + b1)   -> 128 tiles (blk 0-15)
//   layer2: h(64x1024)   = mask ? tanh(xh + mid @ Whh2 + b2) : h -> 256 tiles
// Recurrent weights are scan-invariant: each block stages its private 32-row
// slices of Whh1t (64KB) / Whh2t (32KB) into LDS ONCE, then all 1024 layer
// passes feed the WMMA B-operands from LDS (ds_load_b128) instead of re-
// streaming ~30GB of weight data through L2 over the 512-step scan.
// Dynamic LDS: 96KB/block -> ~3 blocks/WGP, 64 blocks co-resident easily.
// ---------------------------------------------------------------------------
#define SCAN_LDS_W1   (32 * H_)            // bf16 elems: rows [blk*32, +32) of Whh1t
#define SCAN_LDS_W2   (32 * H2_)           // bf16 elems: rows [blk*32, +32) of Whh2t
#define SCAN_LDS_BYTES ((SCAN_LDS_W1 + SCAN_LDS_W2) * 2)

__global__ void __launch_bounds__(256)
scan_kernel(const bf16* __restrict__ Whh1t_f, const float* __restrict__ bhh1_f,
            const bf16* __restrict__ Whh2t_f, const float* __restrict__ bhh2_f,
            const bf16* __restrict__ Whh1t_b, const float* __restrict__ bhh1_b,
            const bf16* __restrict__ Whh2t_b, const float* __restrict__ bhh2_b,
            const float* __restrict__ xh_fwd, const float* __restrict__ xh_bwd,
            float* h_f, float* h_b, bf16* hbf_f, bf16* hbf_b,
            bf16* mid_f, bf16* mid_b,
            const int* __restrict__ lengths,
            float* __restrict__ out_hfinal, float* __restrict__ out_ann,
            unsigned* bar_cnt, unsigned* bar_gen, unsigned nblocks) {
  const int dir  = blockIdx.x & 1;
  const int blk  = blockIdx.x >> 1;          // 0..31 within direction
  const int wave = threadIdx.x >> 5;
  const int lane = threadIdx.x & 31;
  const int wgid = blk * 8 + wave;           // 0..255 per direction

  const bf16*  W1t = dir ? Whh1t_b : Whh1t_f;
  const float* b1  = dir ? bhh1_b  : bhh1_f;
  const bf16*  W2t = dir ? Whh2t_b : Whh2t_f;
  const float* b2  = dir ? bhh2_b  : bhh2_f;
  const float* xh  = dir ? xh_bwd  : xh_fwd;
  float* hbuf = dir ? h_b   : h_f;
  bf16*  hbf  = dir ? hbf_b : hbf_f;
  bf16*  mid  = dir ? mid_b : mid_f;

  extern __shared__ char smem[];
  bf16* lw1 = (bf16*)smem;                       // [32][H_]  (blocks 0-15 only)
  bf16* lw2 = (bf16*)(smem + SCAN_LDS_W1 * 2);   // [32][H2_]

  // ---- one-time LDS staging of this block's weight slices ----
  {
    const uint4* src2 = (const uint4*)(W2t + (size_t)blk * 32 * H2_);
    uint4* dst2 = (uint4*)lw2;
    for (int i = threadIdx.x; i < (32 * H2_) / 8; i += 256) dst2[i] = src2[i];
    if (blk < 16) {   // only these blocks own layer-1 tiles (nt = 2*blk + wave>>2)
      const uint4* src1 = (const uint4*)(W1t + (size_t)blk * 32 * H_);
      uint4* dst1 = (uint4*)lw1;
      for (int i = threadIdx.x; i < (32 * H_) / 8; i += 256) dst1[i] = src1[i];
    }
  }
  __syncthreads();

  const int mt = (wave & 3) << 4;        // tile row base (same for both layers)
  const int lt = (wave >> 2) << 4;       // local B-tile row base in LDS slice

  for (int t = 0; t < S_; ++t) {
    const int s = dir ? (S_ - 1 - t) : t;

    // ---- layer 1: 64x512, K=1024; B from LDS ----
    if (wgid < 128) {
      v8f acc = {};
#pragma unroll 4
      for (int k0 = 0; k0 < H_; k0 += 32) {
        v16bf a = load_frag_k32(hbf, mt, H_, k0);
        v16bf w = load_frag_lds(lw1, lt, H_, k0);
        acc = __builtin_amdgcn_wmma_f32_16x16x32_bf16(
            false, a, false, w, (short)0, acc, false, false);
      }
      const int n  = ((wgid >> 2) << 4) + (lane & 15);
      const int m0 = mt + ((lane >> 4) << 3);
      const float bv = b1[n];
#pragma unroll
      for (int r = 0; r < 8; ++r) {
        float v = acc[r] + bv;
        mid[(size_t)(m0 + r) * H2_ + n] = (bf16)(v > 0.f ? v : 0.f);
      }
    }
    grid_barrier(bar_cnt, bar_gen, nblocks);

    // ---- layer 2: 64x1024, K=512 (256 tiles == 256 waves/dir); B from LDS ----
    {
      v8f acc = {};
#pragma unroll 4
      for (int k0 = 0; k0 < H2_; k0 += 32) {
        v16bf a = load_frag_k32(mid, mt, H2_, k0);
        v16bf w = load_frag_lds(lw2, lt, H2_, k0);
        acc = __builtin_amdgcn_wmma_f32_16x16x32_bf16(
            false, a, false, w, (short)0, acc, false, false);
      }
      const int n  = ((wgid >> 2) << 4) + (lane & 15);
      const int m0 = mt + ((lane >> 4) << 3);
      const float bv = b2[n];
#pragma unroll
      for (int r = 0; r < 8; ++r) {
        const int m = m0 + r;
        const float hprev = hbuf[(size_t)m * H_ + n];
        const float pre = acc[r] + bv + xh[((size_t)s * B_ + m) * H_ + n];
        const float hn = tanhf(pre);
        const float hv = (s < lengths[m]) ? hn : hprev;
        hbuf[(size_t)m * H_ + n] = hv;
        hbf[(size_t)m * H_ + n]  = (bf16)hv;
        out_ann[((size_t)m * S_ + s) * (2 * H_) + (size_t)dir * H_ + n] = hv;
      }
    }
    grid_barrier(bar_cnt, bar_gen, nblocks);
  }

  // h_final = forward h after last step (barrier above made h_f coherent)
  if (dir == 0) {
    for (int i = blk * 256 + threadIdx.x; i < B_ * H_; i += 32 * 256)
      out_hfinal[i] = h_f[i];
  }
}

// ---------------------------------------------------------------------------
extern "C" void kernel_launch(void* const* d_in, const int* in_sizes, int n_in,
                              void* d_out, int out_size, void* d_ws, size_t ws_size,
                              hipStream_t stream) {
  (void)in_sizes; (void)n_in; (void)out_size; (void)ws_size;
  const float* seq    = (const float*)d_in[0];
  const int*   lens   = (const int*)d_in[1];
  const float* Wxh1   = (const float*)d_in[2];
  const float* bxh1   = (const float*)d_in[3];
  const float* Wxh2   = (const float*)d_in[4];
  const float* bxh2   = (const float*)d_in[5];
  const float* Whh1   = (const float*)d_in[6];
  const float* bhh1   = (const float*)d_in[7];
  const float* Whh2   = (const float*)d_in[8];
  const float* bhh2   = (const float*)d_in[9];
  const float* Wxh1_  = (const float*)d_in[10];
  const float* bxh1_  = (const float*)d_in[11];
  const float* Wxh2_  = (const float*)d_in[12];
  const float* bxh2_  = (const float*)d_in[13];
  const float* Whh1_  = (const float*)d_in[14];
  const float* bhh1_  = (const float*)d_in[15];
  const float* Whh2_  = (const float*)d_in[16];
  const float* bhh2_  = (const float*)d_in[17];

  char* ws = (char*)d_ws;
  size_t off = 0;
  auto take = [&](size_t bytes) -> char* {
    char* p = ws + off;
    off = (off + bytes + 255) & ~(size_t)255;
    return p;
  };

  bf16*  seq_bf  = (bf16*)take((size_t)B_ * S_ * E_ * 2);
  bf16*  mid_bf  = (bf16*)take((size_t)B_ * S_ * H2_ * 2);
  float* xh_f    = (float*)take((size_t)S_ * B_ * H_ * 4);
  float* xh_b    = (float*)take((size_t)S_ * B_ * H_ * 4);
  bf16*  wxh1t_f = (bf16*)take((size_t)H2_ * E_ * 2);
  bf16*  wxh2t_f = (bf16*)take((size_t)H_ * H2_ * 2);
  bf16*  whh1t_f = (bf16*)take((size_t)H2_ * H_ * 2);
  bf16*  whh2t_f = (bf16*)take((size_t)H_ * H2_ * 2);
  bf16*  wxh1t_b = (bf16*)take((size_t)H2_ * E_ * 2);
  bf16*  wxh2t_b = (bf16*)take((size_t)H_ * H2_ * 2);
  bf16*  whh1t_b = (bf16*)take((size_t)H2_ * H_ * 2);
  bf16*  whh2t_b = (bf16*)take((size_t)H_ * H2_ * 2);
  float* h_f     = (float*)take((size_t)B_ * H_ * 4);
  float* h_b     = (float*)take((size_t)B_ * H_ * 4);
  bf16*  hbf_f   = (bf16*)take((size_t)B_ * H_ * 2);
  bf16*  hbf_b   = (bf16*)take((size_t)B_ * H_ * 2);
  bf16*  mid_f   = (bf16*)take((size_t)B_ * H2_ * 2);
  bf16*  mid_b   = (bf16*)take((size_t)B_ * H2_ * 2);
  unsigned* bar  = (unsigned*)take(256);

  // --- precision casts / weight transposes (every call; deterministic) ---
  cast_f32_bf16_kernel<<<2048, 256, 0, stream>>>(seq, seq_bf, (size_t)B_ * S_ * E_);
  transpose_cast_kernel<<<512, 256, 0, stream>>>(Wxh1,  wxh1t_f, E_,  H2_);
  transpose_cast_kernel<<<1024, 256, 0, stream>>>(Wxh2,  wxh2t_f, H2_, H_);
  transpose_cast_kernel<<<1024, 256, 0, stream>>>(Whh1,  whh1t_f, H_,  H2_);
  transpose_cast_kernel<<<1024, 256, 0, stream>>>(Whh2,  whh2t_f, H2_, H_);
  transpose_cast_kernel<<<512, 256, 0, stream>>>(Wxh1_, wxh1t_b, E_,  H2_);
  transpose_cast_kernel<<<1024, 256, 0, stream>>>(Wxh2_, wxh2t_b, H2_, H_);
  transpose_cast_kernel<<<1024, 256, 0, stream>>>(Whh1_, whh1t_b, H_,  H2_);
  transpose_cast_kernel<<<1024, 256, 0, stream>>>(Whh2_, whh2t_b, H2_, H_);
  init_state_kernel<<<64, 256, 0, stream>>>(h_f, h_b, hbf_f, hbf_b, bar);

  // --- input MLPs (big WMMA GEMMs), writing xh in (S,B,H) layout ---
  const int M = B_ * S_;
  gemm_bf16_kernel<0, true ><<<dim3(H2_ / 128, M / 64), 256, 0, stream>>>(
      seq_bf, wxh1t_f, bxh1, mid_bf, M, H2_, E_);
  gemm_bf16_kernel<1, false><<<dim3(H_ / 128, M / 64), 256, 0, stream>>>(
      mid_bf, wxh2t_f, bxh2, xh_f, M, H_, H2_);
  gemm_bf16_kernel<0, true ><<<dim3(H2_ / 128, M / 64), 256, 0, stream>>>(
      seq_bf, wxh1t_b, bxh1_, mid_bf, M, H2_, E_);
  gemm_bf16_kernel<1, false><<<dim3(H_ / 128, M / 64), 256, 0, stream>>>(
      mid_bf, wxh2t_b, bxh2_, xh_b, M, H_, H2_);

  // --- persistent bidirectional scan (weights LDS-resident) ---
  float* out_hfinal = (float*)d_out;
  float* out_ann    = (float*)d_out + (size_t)B_ * H_;
  scan_kernel<<<64, 256, SCAN_LDS_BYTES, stream>>>(
      whh1t_f, bhh1, whh2t_f, bhh2,
      whh1t_b, bhh1_, whh2t_b, bhh2_,
      xh_f, xh_b, h_f, h_b, hbf_f, hbf_b, mid_f, mid_b,
      lens, out_hfinal, out_ann, bar, bar + 1, 64u);
}
